// YoloTrainingModel_59261958751040
// MI455X (gfx1250) — compile-verified, matrix operand order checked by reading, not run on hardware
//
#include <hip/hip_runtime.h>
#include <hip/hip_bf16.h>

// ---------------- configuration (mirrors reference) ----------------
#define BB    64      // batch
#define HH    40
#define WW    40
#define AA    5
#define CC    80
#define NBOX  (HH * WW * AA)      // 8000 boxes per image
#define NPAD  8192                // padded for LDS
#define MAXB  100                 // max detections
#define SCORE_THR 0.001f
#define IOU_THR   0.5f
#define REC   85                  // floats per box record (5 + C)

typedef __attribute__((ext_vector_type(16))) _Float16 v16h;
typedef __attribute__((ext_vector_type(8)))  float    v8f;

// =====================================================================
// Kernel 1: decode.  One wave (32 lanes) handles 16 boxes.
//   - stage 16*85 contiguous floats into LDS (coalesced)
//   - per-lane max/argmax over 80 logits
//   - softmax denominator via V_WMMA_F32_16X16X32_F16: D = exp(L-max) x ones
//   - lanes 0..15 finish the scalar decode and store box/score/class
// =====================================================================
__global__ __launch_bounds__(256)
void yolo_decode_kernel(const float* __restrict__ preds,
                        const float* __restrict__ anchors,
                        float4* __restrict__ boxes_out,
                        float*  __restrict__ scores_out,
                        int*    __restrict__ cls_out)
{
    const int lane = threadIdx.x & 31;
    const int wv   = threadIdx.x >> 5;                 // wave in block: 0..7
    const int waveBase = (blockIdx.x * 8 + wv) * 16;   // first box of this wave

    __shared__ float stage[8][16 * REC];               // 43.5 KB
    __shared__ float sums[8][16];

    // ---- stage 16 boxes (contiguous 1360 floats) into LDS ----
    const float* src = preds + (size_t)waveBase * REC;
    if (blockIdx.x + 1 < gridDim.x) {
        // speculative streaming prefetch of the next block's tile
        __builtin_prefetch(src + 8 * 16 * REC, 0, 1);
    }
    #pragma unroll 4
    for (int i = lane; i < 16 * REC; i += 32)
        stage[wv][i] = src[i];
    __syncthreads();

    const int  m  = lane & 15;        // box within wave (lanes l and l+16 share box m)
    const bool hi = (lane >= 16);
    const float* Lg = &stage[wv][m * REC + 5];   // 80 class logits of box m

    // ---- max / argmax over logits (argmax == argmax of softmax scores) ----
    float mx = Lg[0];
    int   am = 0;
    #pragma unroll 8
    for (int k = 1; k < CC; ++k) {
        float v = Lg[k];
        if (v > mx) { mx = v; am = k; }
    }

    // ---- softmax denominator via WMMA: D = E(16x96,f16) * ones(96x16) ----
    v16h ones;
    #pragma unroll
    for (int j = 0; j < 16; ++j) ones[j] = (_Float16)1.0f;

    v8f acc = {};
    #pragma unroll
    for (int chunk = 0; chunk < 3; ++chunk) {
        const int k0 = chunk * 32;
        v16h af;
        // 16-bit A layout: lanes 0-15 hold K = {0..7,16..23}, lanes 16-31 K = {8..15,24..31}
        #pragma unroll
        for (int j = 0; j < 8; ++j) {
            int k = k0 + (hi ? 8 + j : j);
            float e = (k < CC) ? __expf(Lg[k] - mx) : 0.0f;
            af[j] = (_Float16)e;
        }
        #pragma unroll
        for (int j = 0; j < 8; ++j) {
            int k = k0 + (hi ? 24 + j : 16 + j);
            float e = (k < CC) ? __expf(Lg[k] - mx) : 0.0f;
            af[8 + j] = (_Float16)e;
        }
        acc = __builtin_amdgcn_wmma_f32_16x16x32_f16(
                  false, af, false, ones, (short)0, acc, false, false);
    }

    // D layout: lane<16 -> rows 0..7 in acc[0..7]; lane>=16 -> rows 8..15.
    // All 16 columns hold the same row-sum; redundant same-value LDS writes are fine.
    {
        const int rbase = hi ? 8 : 0;
        #pragma unroll
        for (int j = 0; j < 8; ++j)
            sums[wv][rbase + j] = acc[j];
    }
    __syncthreads();

    // ---- scalar decode: one lane per box ----
    if (lane < 16) {
        const float ssum = sums[wv][m];            // sum_k exp(l_k - max)
        const float tx = stage[wv][m * REC + 0];
        const float ty = stage[wv][m * REC + 1];
        const float tw = stage[wv][m * REC + 2];
        const float th = stage[wv][m * REC + 3];
        const float tc = stage[wv][m * REC + 4];

        const int g    = waveBase + m;
        const int a    = g % AA;
        const int cell = g / AA;
        const int gx   = cell % WW;
        const int gy   = (cell / WW) % HH;

        const float sigx = 1.0f / (1.0f + __expf(-tx));
        const float sigy = 1.0f / (1.0f + __expf(-ty));
        const float conf = 1.0f / (1.0f + __expf(-tc));

        const float cx = (sigx + (float)gx) / (float)WW;
        const float cy = (sigy + (float)gy) / (float)HH;
        const float bw = __expf(tw) * anchors[a * 2 + 0] / (float)WW;
        const float bh = __expf(th) * anchors[a * 2 + 1] / (float)HH;

        // max class prob = exp(max-max)/sum = 1/sum  =>  score = conf / sum
        const float score = conf / ssum;
        const float s0    = (score > SCORE_THR) ? score : -1.0f;

        boxes_out[g]  = make_float4(cx - 0.5f * bw, cy - 0.5f * bh,
                                    cx + 0.5f * bw, cy + 0.5f * bh);
        scores_out[g] = s0;
        cls_out[g]    = am;
    }
}

// =====================================================================
// Kernel 2: greedy per-class NMS.  One block per image, 256 threads.
// Scores + classes live in LDS; boxes streamed from L2 as float4.
// =====================================================================
__global__ __launch_bounds__(256)
void yolo_nms_kernel(const float4* __restrict__ boxes,
                     const float*  __restrict__ scores,
                     const int*    __restrict__ cls,
                     float* __restrict__ out_boxes,    // [B,100,4]
                     float* __restrict__ out_scores,   // [B,100]
                     float* __restrict__ out_classes)  // [B,100]
{
    const int img = blockIdx.x;
    const int tid = threadIdx.x;

    __shared__ float         s_lds[NPAD];     // 32 KB
    __shared__ unsigned char c_lds[NPAD];     //  8 KB
    __shared__ float         red_s[256];
    __shared__ int           red_i[256];

    const float4* bimg = boxes  + (size_t)img * NBOX;
    const float*  simg = scores + (size_t)img * NBOX;
    const int*    cimg = cls    + (size_t)img * NBOX;

    for (int i = tid; i < NPAD; i += 256) {
        s_lds[i] = (i < NBOX) ? simg[i] : -1.0f;
        c_lds[i] = (i < NBOX) ? (unsigned char)cimg[i] : 0;
    }
    __syncthreads();

    for (int t = 0; t < MAXB; ++t) {
        // ---- parallel argmax (lowest index on ties, like jnp.argmax) ----
        float bs = -2.0f;
        int   bi = 0;
        for (int i = tid; i < NBOX; i += 256) {
            float v = s_lds[i];
            if (v > bs) { bs = v; bi = i; }
        }
        red_s[tid] = bs;
        red_i[tid] = bi;
        __syncthreads();
        #pragma unroll
        for (int off = 128; off > 0; off >>= 1) {
            if (tid < off) {
                float s2 = red_s[tid + off];
                int   i2 = red_i[tid + off];
                if (s2 > red_s[tid] || (s2 == red_s[tid] && i2 < red_i[tid])) {
                    red_s[tid] = s2;
                    red_i[tid] = i2;
                }
            }
            __syncthreads();
        }
        const float bestS = red_s[0];
        const int   bestI = red_i[0];

        const float4 bb = bimg[bestI];                 // uniform broadcast load
        const int    bc = (int)c_lds[bestI];
        const float  bArea = fmaxf(bb.z - bb.x, 0.0f) * fmaxf(bb.w - bb.y, 0.0f);

        // ---- suppression ----
        for (int i = tid; i < NBOX; i += 256) {
            if (i == bestI) { s_lds[i] = -1.0f; continue; }
            float v = s_lds[i];
            if (v < 0.0f) continue;
            if ((int)c_lds[i] != bc) continue;
            const float4 o = bimg[i];
            const float xx1 = fmaxf(bb.x, o.x), yy1 = fmaxf(bb.y, o.y);
            const float xx2 = fminf(bb.z, o.z), yy2 = fminf(bb.w, o.w);
            const float inter = fmaxf(xx2 - xx1, 0.0f) * fmaxf(yy2 - yy1, 0.0f);
            const float oArea = fmaxf(o.z - o.x, 0.0f) * fmaxf(o.w - o.y, 0.0f);
            const float iou = inter / (oArea + bArea - inter + 1e-8f);
            if (iou > IOU_THR) s_lds[i] = -1.0f;
        }

        if (tid == 0) {
            const bool valid = bestS > 0.0f;
            const int  ob = img * MAXB + t;
            out_boxes[ob * 4 + 0] = valid ? bb.x : 0.0f;
            out_boxes[ob * 4 + 1] = valid ? bb.y : 0.0f;
            out_boxes[ob * 4 + 2] = valid ? bb.z : 0.0f;
            out_boxes[ob * 4 + 3] = valid ? bb.w : 0.0f;
            out_scores[ob]  = valid ? bestS : 0.0f;
            out_classes[ob] = valid ? (float)bc : -1.0f;
        }
        __syncthreads();   // suppression + output visible before next round
    }
}

// =====================================================================
extern "C" void kernel_launch(void* const* d_in, const int* in_sizes, int n_in,
                              void* d_out, int out_size, void* d_ws, size_t ws_size,
                              hipStream_t stream)
{
    const float* preds   = (const float*)d_in[0];   // [64,40,40,425] f32
    const float* anchors = (const float*)d_in[1];   // [5,2] f32

    const size_t NTOT = (size_t)BB * NBOX;          // 512000 boxes

    // workspace layout (16B-aligned slices)
    char*   w      = (char*)d_ws;
    float4* ws_box = (float4*)(w);                           // 8,192,000 B
    float*  ws_sc  = (float*)(w + NTOT * sizeof(float4));    // 2,048,000 B
    int*    ws_cl  = (int*)  (w + NTOT * sizeof(float4) + NTOT * sizeof(float));

    float* out_boxes   = (float*)d_out;                      // [64,100,4]
    float* out_scores  = out_boxes + (size_t)BB * MAXB * 4;  // [64,100]
    float* out_classes = out_scores + (size_t)BB * MAXB;     // [64,100]

    // decode: 512000 boxes / (8 waves * 16 boxes) = 4000 blocks of 256
    yolo_decode_kernel<<<dim3(4000), dim3(256), 0, stream>>>(
        preds, anchors, ws_box, ws_sc, ws_cl);

    // NMS: one block per image
    yolo_nms_kernel<<<dim3(BB), dim3(256), 0, stream>>>(
        ws_box, ws_sc, ws_cl, out_boxes, out_scores, out_classes);
}
